// LinearAttention_10909216931874
// MI455X (gfx1250) — matrix operand order repaired and checked by reading, not compiled, not run on hardware
//
#include <hip/hip_runtime.h>
#include <math.h>

// ---------------------------------------------------------------------------
// Linear attention for MI455X (gfx1250, wave32).
// All GEMM work on v_wmma_f32_16x16x32_bf16 (fp32 accumulate).
// Memory-bound (~1.5GB HBM traffic @ 23.3TB/s): intermediates bf16, WMMA
// operands staged in LDS in native A / B^T layouts (2x ds_load_b128 per
// fragment), transposed LDS staging packs bf16 pairs (ds_store_2addr_b32 +
// v_cvt_pk_bf16_f32), all global stores coalesced through LDS transposes,
// proj A-tile streamed with global_load_async_to_lds_b128 (ASYNCcnt),
// softmax cross-lane reductions via single-instruction ds_swizzle xor.
// Workspace requirement: ~200 MB (layout in kernel_launch).
// ---------------------------------------------------------------------------

typedef __bf16 bf16;
typedef __attribute__((ext_vector_type(16))) __bf16 v16bf;
typedef __attribute__((ext_vector_type(8)))  __bf16 v8bf;
typedef __attribute__((ext_vector_type(2)))  __bf16 v2bf;
typedef __attribute__((ext_vector_type(8)))  float  v8f;

static constexpr int kC     = 256;     // DIM
static constexpr int kN     = 16384;   // 128*128 tokens
static constexpr int kB     = 8;       // batch
static constexpr int kHeads = 8;
static constexpr int kHd    = 32;      // head_dim

__device__ __forceinline__ v8f zero8() {
  v8f z;
#pragma unroll
  for (int i = 0; i < 8; ++i) z[i] = 0.0f;
  return z;
}

__device__ __forceinline__ v2bf pack2f(float a, float b) {
  v2bf r; r[0] = (bf16)a; r[1] = (bf16)b; return r;   // -> v_cvt_pk_bf16_f32
}
__device__ __forceinline__ v2bf pack2b(bf16 a, bf16 b) {
  v2bf r; r[0] = a; r[1] = b; return r;
}

// xor-swizzle across lanes (masks < 16 never cross the 16-lane D-matrix
// group, so this matches a width-16 xor shuffle). Single ds_swizzle_b32.
template <int M>
__device__ __forceinline__ float xorswz(float v) {
  return __int_as_float(
      __builtin_amdgcn_ds_swizzle(__float_as_int(v), (M << 10) | 0x1f));
}

// Assemble a 16-element bf16 fragment from two contiguous 16B chunks.
__device__ __forceinline__ v16bf frag16(const bf16* p0, const bf16* p1) {
  v8bf lo = *(const v8bf*)p0;
  v8bf hi = *(const v8bf*)p1;
  v16bf r;
#pragma unroll
  for (int i = 0; i < 8; ++i) { r[i] = lo[i]; r[i + 8] = hi[i]; }
  return r;
}

__device__ __forceinline__ v8f wmma_bf16(v16bf a, v16bf b, v8f c) {
  return __builtin_amdgcn_wmma_f32_16x16x32_bf16(false, a, false, b,
                                                 (short)0, c, false, false);
}

// ---------------------------------------------------------------------------
// Kernel 1: qkv = x^T @ w_qkv   ([N,256] x [256,768] per batch, bf16 WMMA)
// x is [B][C][N] channel-major -> transpose-staged to LDS (packed b32 stores).
// q columns (<256) get softmax over head_dim fused in registers.
// Output goes through an LDS transpose so global stores are coalesced.
// ---------------------------------------------------------------------------
__global__ __launch_bounds__(256) void qkv_gemm_kernel(
    const float* __restrict__ x,     // [B][256][16384]
    const float* __restrict__ wqkv,  // [256][768]
    bf16* __restrict__ Q,            // [B][N][256]  softmax(q, head_dim)
    bf16* __restrict__ K,            // [B][N][256]  raw k
    bf16* __restrict__ V)            // [B][N][256]  raw v
{
  __shared__ __align__(16) char smem_raw[16384];
  bf16 (*sA)[32]  = (bf16(*)[32])(smem_raw);          // [tok][k]   4KB
  bf16 (*sB)[32]  = (bf16(*)[32])(smem_raw + 4096);   // [col][k]   8KB (B^T)
  bf16 (*sO)[128] = (bf16(*)[128])(smem_raw);         // [tok][col] 16KB alias

  const int b       = blockIdx.z;
  const int tokBase = blockIdx.y * 64;
  const int colBase = blockIdx.x * 128;

  const int tid  = threadIdx.x;
  const int lane = tid & 31;
  const int wave = tid >> 5;
  const int mi   = wave & 3;   // 16-token subtile
  const int nj   = wave >> 2;  // 64-col half
  const int half = lane >> 4;
  const int l16  = lane & 15;

  v8f acc[4];
#pragma unroll
  for (int t = 0; t < 4; ++t) acc[t] = zero8();

  // staging maps (consecutive tids -> consecutive addresses for coalescing)
  const int tgA = tid & 15;         // A: 4-token group
  const int cA0 = (tid >> 4) * 2;   // A: channel pair
  const int jgB = tid & 15;         // B: 8-col group
  const int kB0 = (tid >> 4) * 2;   // B: k pair

  for (int k0 = 0; k0 < kC; k0 += 32) {
    {  // stage A: x[b][k0+c][tokBase+..] -> sA[tok][c], packed channel pairs
      const float* s0 = x + ((size_t)b * kC + (k0 + cA0)) * kN + tokBase + tgA * 4;
      float4 f0 = *(const float4*)(s0);
      float4 f1 = *(const float4*)(s0 + kN);
      if (k0 + 32 < kC) __builtin_prefetch((const void*)(s0 + (size_t)32 * kN), 0, 0);
      const int tr = tgA * 4;
      *(v2bf*)&sA[tr + 0][cA0] = pack2f(f0.x, f1.x);
      *(v2bf*)&sA[tr + 1][cA0] = pack2f(f0.y, f1.y);
      *(v2bf*)&sA[tr + 2][cA0] = pack2f(f0.z, f1.z);
      *(v2bf*)&sA[tr + 3][cA0] = pack2f(f0.w, f1.w);
    }
    {  // stage B^T: wqkv[k0+k][colBase+..] -> sB[col][k], packed k pairs
      const float* s0 = wqkv + (size_t)(k0 + kB0) * (3 * kC) + colBase + jgB * 8;
      float4 r00 = *(const float4*)(s0);
      float4 r01 = *(const float4*)(s0 + 4);
      float4 r10 = *(const float4*)(s0 + 3 * kC);
      float4 r11 = *(const float4*)(s0 + 3 * kC + 4);
      const int jc = jgB * 8;
      *(v2bf*)&sB[jc + 0][kB0] = pack2f(r00.x, r10.x);
      *(v2bf*)&sB[jc + 1][kB0] = pack2f(r00.y, r10.y);
      *(v2bf*)&sB[jc + 2][kB0] = pack2f(r00.z, r10.z);
      *(v2bf*)&sB[jc + 3][kB0] = pack2f(r00.w, r10.w);
      *(v2bf*)&sB[jc + 4][kB0] = pack2f(r01.x, r11.x);
      *(v2bf*)&sB[jc + 5][kB0] = pack2f(r01.y, r11.y);
      *(v2bf*)&sB[jc + 6][kB0] = pack2f(r01.z, r11.z);
      *(v2bf*)&sB[jc + 7][kB0] = pack2f(r01.w, r11.w);
    }
    __syncthreads();

    // A frag: row m = 16*mi + l16 ; K elems 16h..16h+15 contiguous
    const bf16* ap = &sA[16 * mi + l16][16 * half];
    v16bf af = frag16(ap, ap + 8);
#pragma unroll
    for (int t = 0; t < 4; ++t) {
      // B frag: col = 64*nj + 16*t + l16 ; K elems 16h..16h+15 contiguous
      const bf16* bp = &sB[64 * nj + 16 * t + l16][16 * half];
      v16bf bfr = frag16(bp, bp + 8);
      acc[t] = wmma_bf16(af, bfr, acc[t]);
    }
    __syncthreads();
  }

  // Fused q-softmax over head_dim: wave covers 64 aligned cols = 2 full heads.
  // D layout: VGPR i -> row (i + 8*half); head's 32 cols = 2 acc tiles across
  // lanes 0..15 / 16..31; xor-swizzle masks stay inside each 16-lane group.
  if (colBase < 256) {
#pragma unroll
    for (int p = 0; p < 2; ++p) {
#pragma unroll
      for (int i = 0; i < 8; ++i) {
        float v0 = acc[2 * p][i], v1 = acc[2 * p + 1][i];
        float mx = fmaxf(v0, v1);
        mx = fmaxf(mx, xorswz<1>(mx));
        mx = fmaxf(mx, xorswz<2>(mx));
        mx = fmaxf(mx, xorswz<4>(mx));
        mx = fmaxf(mx, xorswz<8>(mx));
        float e0 = __expf(v0 - mx), e1 = __expf(v1 - mx);
        float s = e0 + e1;
        s += xorswz<1>(s);
        s += xorswz<2>(s);
        s += xorswz<4>(s);
        s += xorswz<8>(s);
        float inv = 1.0f / s;
        acc[2 * p][i]     = e0 * inv;
        acc[2 * p + 1][i] = e1 * inv;
      }
    }
  }

  // D tiles -> LDS transpose -> coalesced bf16 stores (16B per lane per row)
#pragma unroll
  for (int t = 0; t < 4; ++t) {
    const int lcol = 64 * nj + 16 * t + l16;
#pragma unroll
    for (int i = 0; i < 8; ++i)
      sO[16 * mi + 8 * half + i][lcol] = (bf16)acc[t][i];
  }
  __syncthreads();

  const size_t nbase = (size_t)b * kN;
  bf16* dst; int c0;
  if (colBase < 256)      { dst = Q; c0 = colBase; }
  else if (colBase < 512) { dst = K; c0 = colBase - 256; }
  else                    { dst = V; c0 = colBase - 512; }
  const int row = tid >> 2, seg = tid & 3;  // 64 rows x 4 segs of 32 cols
  bf16* drow = dst + (nbase + tokBase + row) * kC + c0 + seg * 32;
#pragma unroll
  for (int q = 0; q < 4; ++q)
    *(v8bf*)(drow + q * 8) = *(const v8bf*)&sO[row][seg * 32 + q * 8];
}

// ---------------------------------------------------------------------------
// Kernel 2/3: k-softmax denominators: ksum[b][c] = sum_n exp(k[b][n][c]).
// (exp without max is safe: k ~ N(0,1).) Deterministic 2-stage reduction.
// ---------------------------------------------------------------------------
__global__ __launch_bounds__(256) void ksum_partial_kernel(
    const bf16* __restrict__ K, float* __restrict__ part) {
  const int b = blockIdx.y, ch = blockIdx.x;  // 128 chunks of 128 tokens
  const int col = threadIdx.x;
  const bf16* p = K + ((size_t)b * kN + (size_t)ch * 128) * kC + col;
  float s = 0.0f;
  for (int t = 0; t < 128; ++t) s += __expf((float)p[(size_t)t * kC]);
  part[((size_t)b * 128 + ch) * kC + col] = s;
}

__global__ __launch_bounds__(256) void ksum_reduce_kernel(
    const float* __restrict__ part, float* __restrict__ ksum) {
  const int b = blockIdx.x, col = threadIdx.x;
  float s = 0.0f;
  for (int ch = 0; ch < 128; ++ch) s += part[((size_t)b * 128 + ch) * kC + col];
  ksum[b * kC + col] = s;
}

// ---------------------------------------------------------------------------
// Kernel 4: context partials: cpart[bh][chunk] = exp(K_h)^T @ V_h over a
// 2048-token chunk ([32 x 2048] x [2048 x 32], bf16 WMMA, split-K).
// exp() applied while transpose-staging (packed token pairs -> ds_store_b32).
// ---------------------------------------------------------------------------
__global__ __launch_bounds__(256) void ctx_partial_kernel(
    const bf16* __restrict__ K, const bf16* __restrict__ V,
    float* __restrict__ cpart) {
  __shared__ __align__(16) char smem[32768];
  bf16 (*sK)[256] = (bf16(*)[256])(smem);            // [d][tok] exp(K^T) 16KB
  bf16 (*sV)[256] = (bf16(*)[256])(smem + 16384);    // [e][tok] V^T      16KB
  float* red = (float*)smem;                         // alias after loop  32KB

  const int bh = blockIdx.x;           // b*8 + h
  const int b = bh >> 3, h = bh & 7;
  const int chunk = blockIdx.y;        // 2048 tokens per chunk
  const int tid = threadIdx.x, lane = tid & 31, wave = tid >> 5;
  const int half = lane >> 4, l16 = lane & 15;

  v8f acc[4];
#pragma unroll
  for (int t = 0; t < 4; ++t) acc[t] = zero8();

  const size_t rowBase = (size_t)b * kN * kC + h * kHd;
  const int p  = tid & 127;   // token pair index
  const int dh = tid >> 7;    // d half: 0..15 or 16..31

  for (int slab = 0; slab < 8; ++slab) {           // 8 x 256-token slabs
    const int tok0 = chunk * 2048 + slab * 256;
    {  // stage two tokens (2p, 2p+1) for 16 d-channels, packed b32 stores
      const bf16* kp = K + rowBase + (size_t)(tok0 + 2 * p) * kC + 16 * dh;
      const bf16* vp = V + rowBase + (size_t)(tok0 + 2 * p) * kC + 16 * dh;
      v8bf k0a = ((const v8bf*)kp)[0],        k0b = ((const v8bf*)kp)[1];
      v8bf k1a = ((const v8bf*)(kp + kC))[0], k1b = ((const v8bf*)(kp + kC))[1];
      v8bf v0a = ((const v8bf*)vp)[0],        v0b = ((const v8bf*)vp)[1];
      v8bf v1a = ((const v8bf*)(vp + kC))[0], v1b = ((const v8bf*)(vp + kC))[1];
#pragma unroll
      for (int j = 0; j < 8; ++j) {
        const int d = 16 * dh + j;
        *(v2bf*)&sK[d][2 * p] = pack2f(__expf((float)k0a[j]), __expf((float)k1a[j]));
        *(v2bf*)&sV[d][2 * p] = pack2b(v0a[j], v1a[j]);
      }
#pragma unroll
      for (int j = 0; j < 8; ++j) {
        const int d = 16 * dh + 8 + j;
        *(v2bf*)&sK[d][2 * p] = pack2f(__expf((float)k0b[j]), __expf((float)k1b[j]));
        *(v2bf*)&sV[d][2 * p] = pack2b(v0b[j], v1b[j]);
      }
    }
    __syncthreads();
    const int tw = wave * 32;  // each wave: its own 32-token K-step
#pragma unroll
    for (int mt = 0; mt < 2; ++mt) {
      const bf16* ap = &sK[16 * mt + l16][tw + 16 * half];
      v16bf af = frag16(ap, ap + 8);
#pragma unroll
      for (int nt = 0; nt < 2; ++nt) {
        const bf16* bp = &sV[16 * nt + l16][tw + 16 * half];
        v16bf bfr = frag16(bp, bp + 8);
        acc[mt * 2 + nt] = wmma_bf16(af, bfr, acc[mt * 2 + nt]);
      }
    }
    __syncthreads();
  }

  // cross-wave reduction of the 32x32 tile through LDS (deterministic)
#pragma unroll
  for (int mt = 0; mt < 2; ++mt)
#pragma unroll
    for (int nt = 0; nt < 2; ++nt)
#pragma unroll
      for (int i = 0; i < 8; ++i) {
        const int d = 16 * mt + 8 * half + i;
        const int e = 16 * nt + l16;
        red[wave * 1024 + d * 32 + e] = acc[mt * 2 + nt][i];
      }
  __syncthreads();
#pragma unroll
  for (int r = 0; r < 4; ++r) {
    const int idx = tid * 4 + r;  // 0..1023 = d*32+e
    float s = 0.0f;
#pragma unroll
    for (int w = 0; w < 8; ++w) s += red[w * 1024 + idx];
    cpart[((size_t)bh * 8 + chunk) * 1024 + idx] = s;
  }
}

// Kernel 5: reduce chunk partials + normalize rows by ksum -> ctx[bh][d][e]
__global__ __launch_bounds__(256) void ctx_reduce_kernel(
    const float* __restrict__ cpart, const float* __restrict__ ksum,
    float* __restrict__ ctx) {
  const int bh = blockIdx.x, tid = threadIdx.x;
  const int b = bh >> 3, h = bh & 7;
#pragma unroll
  for (int r = 0; r < 4; ++r) {
    const int idx = tid * 4 + r;
    const int d = idx >> 5;
    float s = 0.0f;
#pragma unroll
    for (int ch = 0; ch < 8; ++ch) s += cpart[((size_t)bh * 8 + ch) * 1024 + idx];
    ctx[(size_t)bh * 1024 + idx] = s / ksum[b * kC + h * kHd + d];
  }
}

// ---------------------------------------------------------------------------
// Kernel 6: attn[b][n][h*32+e] = q_soft @ ctx[bh]  (WMMA, K=32).
// 4 waves / 64 tokens per block; per-wave LDS staging so global stores are
// 16B-contiguous per lane (512B per row across the wave).
// ---------------------------------------------------------------------------
__global__ __launch_bounds__(128) void attn_kernel(
    const bf16* __restrict__ Q, const float* __restrict__ ctx,
    bf16* __restrict__ A) {
  __shared__ bf16 sC[8][32][32];     // [h][e][d] B^T, 16KB
  __shared__ bf16 sO[4][16][256];    // per-wave [tok][col] staging, 32KB
  const int b = blockIdx.y;
  const int tid = threadIdx.x, lane = tid & 31, wave = tid >> 5;
  const int half = lane >> 4, l16 = lane & 15;

  for (int r = tid; r < 8 * 1024; r += 128) {
    const int h = r >> 10, rem = r & 1023, d = rem >> 5, e = rem & 31;
    sC[h][e][d] = (bf16)ctx[((size_t)(b * 8 + h)) * 1024 + rem];
  }
  __syncthreads();

  const int tokBase = blockIdx.x * 64 + wave * 16;
  const size_t nbase = (size_t)b * kN;
#pragma unroll
  for (int h = 0; h < 8; ++h) {
    const bf16* qp = Q + (nbase + tokBase + l16) * kC + h * kHd + 16 * half;
    v16bf af = frag16(qp, qp + 8);
#pragma unroll
    for (int nt = 0; nt < 2; ++nt) {
      const bf16* bp = &sC[h][16 * nt + l16][16 * half];
      v16bf bfr = frag16(bp, bp + 8);
      v8f c = zero8();
      c = wmma_bf16(af, bfr, c);
      const int col = h * kHd + 16 * nt + l16;
#pragma unroll
      for (int i = 0; i < 8; ++i)
        sO[wave][8 * half + i][col] = (bf16)c[i];
    }
  }
  // per-wave staging: same-wave LDS RAW, no block barrier needed
#pragma unroll
  for (int r = 0; r < 16; ++r)
    *(v8bf*)(A + (nbase + tokBase + r) * kC + lane * 8) =
        *(const v8bf*)&sO[wave][r][lane * 8];
}

// ---------------------------------------------------------------------------
// Kernel 7: out[b][c][n] = (attn @ w_proj + b_proj)^T.
// A tile streamed with global_load_async_to_lds_b128 (ASYNCcnt path);
// output transposed through LDS for contiguous channel-major stores.
// ---------------------------------------------------------------------------
__global__ __launch_bounds__(256) void proj_kernel(
    const bf16* __restrict__ A, const float* __restrict__ wproj,
    const float* __restrict__ bias, float* __restrict__ out) {
  __shared__ __align__(16) char smem[16384];
  bf16 (*sA)[32] = (bf16(*)[32])(smem);          // [tok][k]  4KB
  bf16 (*sB)[32] = (bf16(*)[32])(smem + 4096);   // [col][k]  4KB (B^T)
  float (*sO)[64] = (float(*)[64])(smem);        // [col][tok] alias, 16KB

  const int b = blockIdx.z;
  const int tokBase = blockIdx.y * 64;
  const int colBase = blockIdx.x * 64;
  const int tid = threadIdx.x, lane = tid & 31, wave = tid >> 5;
  const int half = lane >> 4, l16 = lane & 15;
  const int mi = wave & 3, nj = wave >> 2;

  v8f acc[2];
  acc[0] = zero8(); acc[1] = zero8();

  const size_t abase = (size_t)b * kN * kC;
  const int rowS = tid >> 2, segS = tid & 3;       // A staging: 16B per thread
  const int jgP = tid & 15, kP0 = (tid >> 4) * 2;  // B staging: 4 cols, k pair

  // per-lane LDS byte address for the async A-tile copy (flat addr low 32b)
  const unsigned ldsA = (unsigned)(uintptr_t)&sA[rowS][segS * 8];

  for (int k0 = 0; k0 < kC; k0 += 32) {
    {  // async DMA: attn tile (bf16, no transform) global -> LDS, 16B/lane
      const bf16* src = A + abase + (size_t)(tokBase + rowS) * kC + k0 + segS * 8;
      unsigned long long gaddr = (unsigned long long)(uintptr_t)src;
      asm volatile("global_load_async_to_lds_b128 %0, %1, off"
                   :: "v"(ldsA), "v"(gaddr) : "memory");
      if (k0 + 32 < kC) __builtin_prefetch((const void*)(src + 32), 0, 0);
    }
    {  // stage B^T: wproj[k0+k][colBase+..] -> sB[col][k], packed k pairs
      const float* s0 = wproj + (size_t)(k0 + kP0) * kC + colBase + jgP * 4;
      float4 r0 = *(const float4*)(s0);
      float4 r1 = *(const float4*)(s0 + kC);
      const int jc = jgP * 4;
      *(v2bf*)&sB[jc + 0][kP0] = pack2f(r0.x, r1.x);
      *(v2bf*)&sB[jc + 1][kP0] = pack2f(r0.y, r1.y);
      *(v2bf*)&sB[jc + 2][kP0] = pack2f(r0.z, r1.z);
      *(v2bf*)&sB[jc + 3][kP0] = pack2f(r0.w, r1.w);
    }
    asm volatile("s_wait_asynccnt 0" ::: "memory");
    __syncthreads();

    const bf16* ap = &sA[16 * mi + l16][16 * half];
    v16bf af = frag16(ap, ap + 8);
#pragma unroll
    for (int t = 0; t < 2; ++t) {
      const bf16* bp = &sB[32 * nj + 16 * t + l16][16 * half];
      v16bf bfr = frag16(bp, bp + 8);
      acc[t] = wmma_bf16(af, bfr, acc[t]);
    }
    __syncthreads();
  }

  // bias + transpose through LDS (sO aliases sA/sB; loop ended on a barrier)
#pragma unroll
  for (int t = 0; t < 2; ++t) {
    const int lcol = 32 * nj + 16 * t + l16;
    const float bv = bias[colBase + lcol];
#pragma unroll
    for (int i = 0; i < 8; ++i)
      sO[lcol][16 * mi + 8 * half + i] = acc[t][i] + bv;
  }
  __syncthreads();

  const int col = tid >> 2, part = tid & 3;  // 64 cols x 4 parts of 16 tokens
  float* dst = out + ((size_t)b * kC + colBase + col) * kN + tokBase + part * 16;
#pragma unroll
  for (int q = 0; q < 4; ++q)
    *(float4*)(dst + q * 4) = *(float4*)&sO[col][part * 16 + q * 4];
}

// ---------------------------------------------------------------------------
extern "C" void kernel_launch(void* const* d_in, const int* in_sizes, int n_in,
                              void* d_out, int out_size, void* d_ws, size_t ws_size,
                              hipStream_t stream) {
  (void)in_sizes; (void)n_in; (void)out_size; (void)ws_size;
  const float* x     = (const float*)d_in[0];  // [8][256][128*128]
  const float* wqkv  = (const float*)d_in[1];  // [256][768]
  const float* wproj = (const float*)d_in[2];  // [256][256]
  const float* bproj = (const float*)d_in[3];  // [256]
  float* out = (float*)d_out;                  // [8][256][128*128]

  // workspace layout (bytes); ATTN aliases V (V dead after ctx_partial)
  char* ws = (char*)d_ws;
  const size_t SZ = (size_t)kB * kN * kC * sizeof(bf16);  // 64 MB
  bf16*  Q     = (bf16*)(ws);
  bf16*  K     = (bf16*)(ws + SZ);
  bf16*  V     = (bf16*)(ws + 2 * SZ);
  bf16*  ATTN  = V;                                        // alias
  float* kpart = (float*)(ws + 3 * SZ);                    // 8*128*256 f32 (1MB)
  float* ksum  = (float*)(ws + 3 * SZ + (1u << 20));       // 8*256 f32
  float* cpart = (float*)(ws + 3 * SZ + (1u << 20) + (1u << 16));  // 64*8*1024 f32
  float* ctx   = (float*)(ws + 3 * SZ + (1u << 20) + (1u << 16) + (2u << 20));

  qkv_gemm_kernel<<<dim3(6, kN / 64, kB), 256, 0, stream>>>(x, wqkv, Q, K, V);
  ksum_partial_kernel<<<dim3(128, kB), 256, 0, stream>>>(K, kpart);
  ksum_reduce_kernel<<<kB, 256, 0, stream>>>(kpart, ksum);
  ctx_partial_kernel<<<dim3(kB * kHeads, 8), 256, 0, stream>>>(K, V, cpart);
  ctx_reduce_kernel<<<kB * kHeads, 256, 0, stream>>>(cpart, ksum, ctx);
  attn_kernel<<<dim3(kN / 64, kB), 128, 0, stream>>>(Q, ctx, ATTN);
  proj_kernel<<<dim3(kC / 64, kN / 64, kB), 256, 0, stream>>>(ATTN, wproj, bproj, out);
}
